// GRUModel_29600914604716
// MI455X (gfx1250) — compile-verified, matrix operand order checked by reading, not run on hardware
//
#include <hip/hip_runtime.h>

// Problem constants (match reference)
#define VOCABSZ 3
#define EMBSZ   64
#define HIDSZ   4
#define OUTSZ   2
#define BSZ     256
#define TLEN    4096

typedef _Float16 v16h __attribute__((ext_vector_type(16)));
typedef float    v8f  __attribute__((ext_vector_type(8)));

// CDNA5 hardware tanh (TRANS32 op). Builtin if available, else inline asm.
__device__ __forceinline__ float hw_tanh(float x) {
#if __has_builtin(__builtin_amdgcn_tanhf)
    return __builtin_amdgcn_tanhf(x);
#else
    float r;
    asm volatile("v_tanh_f32 %0, %1" : "=v"(r) : "v"(x));
    return r;
#endif
}
// sigmoid(x) = 0.5*tanh(x/2) + 0.5  -> mul + v_tanh + fma (3 VALU)
__device__ __forceinline__ float fast_sigmoid(float x) {
    return fmaf(0.5f, hw_tanh(0.5f * x), 0.5f);
}

__device__ __forceinline__ float dot64v(const float* __restrict__ w,
                                        const float* __restrict__ e) {
    const float4* w4 = (const float4*)w;
    const float4* e4 = (const float4*)e;
    float s = 0.0f;
#pragma unroll 4
    for (int i = 0; i < 16; ++i) {
        float4 a = w4[i], b = e4[i];
        s = fmaf(a.x, b.x, s); s = fmaf(a.y, b.y, s);
        s = fmaf(a.z, b.z, s); s = fmaf(a.w, b.w, s);
    }
    return s;
}

// One workgroup = one wave32; lane -> one batch row. 8 blocks total.
__global__ __launch_bounds__(32, 1)
void bigru_fused_kernel(const int*   __restrict__ x,      // [B,T] tokens (int32 view)
                        const float* __restrict__ emb,    // [3,64]
                        const float* __restrict__ Wih_f,  // [12,64]
                        const float* __restrict__ Whh_f,  // [12,4]
                        const float* __restrict__ bih_f,  // [12]
                        const float* __restrict__ bhh_f,  // [12]
                        const float* __restrict__ Wih_b,
                        const float* __restrict__ Whh_b,  // unused: h0=0 -> only bias matters
                        const float* __restrict__ bih_b,
                        const float* __restrict__ bhh_b,
                        const float* __restrict__ Wout,   // [2,8]
                        const float* __restrict__ bout,   // [2]
                        float*       __restrict__ out)    // [B,2]
{
    const int lane = threadIdx.x;
    const int brow = blockIdx.x * 32 + lane;
    (void)Whh_b;

    // LDS layout (floats):
    //   [0..35]   forward token table tab[v][g] (biases folded)  -- live all loop
    //   [36..71]  backward raw dots (consumed at startup)
    //   [48..175] reused as f16 hbuf[32][8] for the WMMA epilogue
    __shared__ alignas(16) float smem[176];

    // ------------------------------------------------------------------
    // Cooperative startup: 72 dot products (one per lane, 3 rounds).
    //   j < 36 : forward  W_ih_f[g]·emb[v]   -> smem[v*12+g]
    //   j >= 36: backward W_ih_b[g]·emb[v]   -> smem[36+v*12+g]
    // ------------------------------------------------------------------
    for (int j = lane; j < 72; j += 32) {
        const int idx = (j < 36) ? j : (j - 36);
        const int v = idx / 12, g = idx % 12;
        const float* W = (j < 36) ? Wih_f : Wih_b;
        smem[j] = dot64v(W + g * EMBSZ, emb + v * EMBSZ);
    }
    __syncthreads();

    // Backward direction: reference only consumes out_b[-1] = ONE GRU step
    // from h0=0 on token x[:,T-1] -> 3-entry lookup table (per-lane regs).
    float hbT[VOCABSZ][HIDSZ];
#pragma unroll
    for (int v = 0; v < VOCABSZ; ++v) {
        float xb[12];
#pragma unroll
        for (int j = 0; j < 3; ++j) {
            float4 q = ((const float4*)(smem + 36))[v * 3 + j];
            xb[j*4+0] = q.x; xb[j*4+1] = q.y; xb[j*4+2] = q.z; xb[j*4+3] = q.w;
        }
#pragma unroll
        for (int i = 0; i < HIDSZ; ++i) {
            float r = fast_sigmoid(xb[i]     + bih_b[i]     + bhh_b[i]);
            float z = fast_sigmoid(xb[4 + i] + bih_b[4 + i] + bhh_b[4 + i]);
            float n = hw_tanh(fmaf(r, bhh_b[8 + i], xb[8 + i] + bih_b[8 + i]));
            hbT[v][i] = (1.0f - z) * n;
        }
    }

    // Fold forward biases into the LDS table (r,z also get b_hh folded).
    for (int j = lane; j < 36; j += 32) {
        const int g = j % 12;
        smem[j] += bih_f[g] + ((g < 8) ? bhh_f[g] : 0.0f);
    }
    __syncthreads();

    // Hidden-to-hidden weights: uniform loads -> SGPRs.
    float Whh[12][HIDSZ];
#pragma unroll
    for (int g = 0; g < 12; ++g)
#pragma unroll
        for (int k = 0; k < HIDSZ; ++k) Whh[g][k] = Whh_f[g * HIDSZ + k];
    float bhhn[HIDSZ];
#pragma unroll
    for (int i = 0; i < HIDSZ; ++i) bhhn[i] = bhh_f[8 + i];

    // ------------------------------------------------------------------
    // Forward scan: 4096 sequential GRU steps, one batch row per lane.
    // Token-table reads are 3x ds_load_b128 per step, issued for a whole 4-step group
    // before its first step (tokens known in advance -> off the serial chain).
    // ------------------------------------------------------------------
    float h[HIDSZ] = {0.f, 0.f, 0.f, 0.f};

    auto step = [&](const float4& xr4, const float4& xz4, const float4& xn4) {
        float pr[4] = {xr4.x, xr4.y, xr4.z, xr4.w};
        float pz[4] = {xz4.x, xz4.y, xz4.z, xz4.w};
        float pn[4] = {xn4.x, xn4.y, xn4.z, xn4.w};
#pragma unroll
        for (int i = 0; i < 4; ++i)
#pragma unroll
            for (int k = 0; k < 4; ++k) {
                pr[i] = fmaf(Whh[i][k],     h[k], pr[i]);
                pz[i] = fmaf(Whh[4 + i][k], h[k], pz[i]);
            }
        float hn[4];
#pragma unroll
        for (int i = 0; i < 4; ++i) {
            float a = bhhn[i];
#pragma unroll
            for (int k = 0; k < 4; ++k) a = fmaf(Whh[8 + i][k], h[k], a);
            hn[i] = a;
        }
#pragma unroll
        for (int i = 0; i < 4; ++i) {
            float r = fast_sigmoid(pr[i]);
            float z = fast_sigmoid(pz[i]);
            float n = hw_tanh(fmaf(r, hn[i], pn[i]));
            h[i] = fmaf(z, h[i] - n, n);            // (1-z)*n + z*h
        }
    };

    const float4* tabv = (const float4*)smem;       // tab[v] = tabv[v*3 + {0,1,2}]
    const int* tp = x + (size_t)brow * TLEN;
    int4 cur = *reinterpret_cast<const int4*>(tp);

    for (int t = 0; t + 4 < TLEN; t += 4) {
        int4 nxt = *reinterpret_cast<const int4*>(tp + t + 4);   // double-buffer
        __builtin_prefetch(tp + t + 64, 0, 1);                   // global_prefetch_b8
        const int toks[4] = {cur.x, cur.y, cur.z, cur.w};
        float4 xr[4], xz[4], xn[4];
#pragma unroll
        for (int j = 0; j < 4; ++j) {                            // 12 ds_load_b128, early
            const float4* p = tabv + toks[j] * 3;
            xr[j] = p[0]; xz[j] = p[1]; xn[j] = p[2];
        }
#pragma unroll
        for (int j = 0; j < 4; ++j) step(xr[j], xz[j], xn[j]);
        cur = nxt;
    }
    {   // tail group (tokens 4092..4095)
        const int toks[4] = {cur.x, cur.y, cur.z, cur.w};
        float4 xr[4], xz[4], xn[4];
#pragma unroll
        for (int j = 0; j < 4; ++j) {
            const float4* p = tabv + toks[j] * 3;
            xr[j] = p[0]; xz[j] = p[1]; xn[j] = p[2];
        }
#pragma unroll
        for (int j = 0; j < 4; ++j) step(xr[j], xz[j], xn[j]);
    }
    const int lastTok = cur.w;

    // Backward hidden for this row: table select by last token (branchless).
    const bool l0 = (lastTok == 0), l1 = (lastTok == 1);
    float hb[HIDSZ];
#pragma unroll
    for (int i = 0; i < HIDSZ; ++i)
        hb[i] = l0 ? hbT[0][i] : (l1 ? hbT[1][i] : hbT[2][i]);

    // ------------------------------------------------------------------
    // Fused output projection with WMMA:
    //   out[32,2] = concat(hf,hb)[32,8] @ Wout^T[8,2] + bout
    // 16-bit A layout: lanes 0-15 hold K=0..7 (valid), lanes 16-31 K=8..15 (zero).
    // All LDS row reads are unconditional b128 loads, masked with cndmask.
    // ------------------------------------------------------------------
    _Float16* hbuf = (_Float16*)(smem + 48);
#pragma unroll
    for (int i = 0; i < HIDSZ; ++i) {
        hbuf[lane * 8 + i]         = (_Float16)h[i];
        hbuf[lane * 8 + HIDSZ + i] = (_Float16)hb[i];
    }
    __syncthreads();

    const int  n       = lane & 15;
    const bool lowHalf = (lane < 16);
    struct U8 { uint4 lo, hi; };

    // B[k][n] = Wout[n][k] (k<8, n<2), zero elsewhere. Uniform values, lane select.
    unsigned bdw[4];
#pragma unroll
    for (int j = 0; j < 4; ++j) {
        union { unsigned u; _Float16 f[2]; } cv;
        float e0 = (n == 0) ? Wout[2*j]     : ((n == 1) ? Wout[8 + 2*j]     : 0.0f);
        float e1 = (n == 0) ? Wout[2*j + 1] : ((n == 1) ? Wout[8 + 2*j + 1] : 0.0f);
        cv.f[0] = (_Float16)e0; cv.f[1] = (_Float16)e1;
        bdw[j] = lowHalf ? cv.u : 0u;
    }
    U8 bu; bu.lo = make_uint4(bdw[0], bdw[1], bdw[2], bdw[3]);
    bu.hi = make_uint4(0u, 0u, 0u, 0u);
    v16h bm = __builtin_bit_cast(v16h, bu);

    // C[m][n] = bout[n] broadcast down all rows.
    const float bias = (n == 0) ? bout[0] : ((n == 1) ? bout[1] : 0.0f);
    v8f c;
#pragma unroll
    for (int j = 0; j < 8; ++j) c[j] = bias;

#pragma unroll
    for (int tile = 0; tile < 2; ++tile) {
        const int row = tile * 16 + n;
        uint4 rv = ((const uint4*)(smem + 48))[row];      // one ds_load_b128
        U8 au;
        au.lo = make_uint4(lowHalf ? rv.x : 0u, lowHalf ? rv.y : 0u,
                           lowHalf ? rv.z : 0u, lowHalf ? rv.w : 0u);
        au.hi = make_uint4(0u, 0u, 0u, 0u);
        v16h a = __builtin_bit_cast(v16h, au);

        v8f d = __builtin_amdgcn_wmma_f32_16x16x32_f16(
            /*neg_a=*/false, a, /*neg_b=*/false, bm,
            /*c_mod=*/(short)0, c, /*reuse_a=*/false, /*reuse_b=*/false);

        if (n < OUTSZ) {
            const int mbase = blockIdx.x * 32 + tile * 16 + (lowHalf ? 0 : 8);
#pragma unroll
            for (int j = 0; j < 8; ++j)
                out[(mbase + j) * OUTSZ + n] = d[j];
        }
    }
}

extern "C" void kernel_launch(void* const* d_in, const int* in_sizes, int n_in,
                              void* d_out, int out_size, void* d_ws, size_t ws_size,
                              hipStream_t stream) {
    (void)in_sizes; (void)n_in; (void)d_ws; (void)ws_size; (void)out_size;
    const int*   x     = (const int*)d_in[0];
    const float* emb   = (const float*)d_in[1];
    const float* Wih_f = (const float*)d_in[2];
    const float* Whh_f = (const float*)d_in[3];
    const float* bih_f = (const float*)d_in[4];
    const float* bhh_f = (const float*)d_in[5];
    const float* Wih_b = (const float*)d_in[6];
    const float* Whh_b = (const float*)d_in[7];
    const float* bih_b = (const float*)d_in[8];
    const float* bhh_b = (const float*)d_in[9];
    const float* Wout  = (const float*)d_in[10];
    const float* bout  = (const float*)d_in[11];

    bigru_fused_kernel<<<BSZ / 32, 32, 0, stream>>>(
        x, emb, Wih_f, Whh_f, bih_f, bhh_f,
        Wih_b, Whh_b, bih_b, bhh_b, Wout, bout, (float*)d_out);
}
